// MultiHeadLatentAttention_10213432229912
// MI455X (gfx1250) — compile-verified, matrix operand order checked by reading, not run on hardware
//
#include <hip/hip_runtime.h>
#include <stdint.h>

// ---------------- problem constants ----------------
#define BB 2
#define SS 2048
#define DDIM 2048
#define NH 16
#define DL 512
#define HD 128   // DDIM / NH

// ---------------- WMMA plumbing ----------------
typedef __attribute__((ext_vector_type(16))) __bf16 v16bf;
typedef __attribute__((ext_vector_type(8)))  float  v8f;
typedef __attribute__((ext_vector_type(4)))  unsigned int u32x4;
typedef __attribute__((__vector_size__(16))) int v4i;
typedef __attribute__((address_space(1))) v4i gv4i;   // global v4i
typedef __attribute__((address_space(3))) v4i lv4i;   // LDS v4i

union Frag16 {
  v16bf v;
  u32x4 q[2];
};

__device__ __forceinline__ unsigned short f2bf(float f) {
  unsigned int u = __float_as_uint(f);
  u = (u + 0x7FFFu + ((u >> 16) & 1u)) >> 16;
  return (unsigned short)u;
}

__device__ __forceinline__ v8f wmma_bf16(const Frag16& a, const Frag16& b, v8f c) {
  // D = A(16x32 bf16) * B(32x16 bf16) + C(16x16 f32)
  return __builtin_amdgcn_wmma_f32_16x16x32_bf16(false, a.v, false, b.v,
                                                 (short)0, c, false, false);
}

// A/B fragment per-lane packing (16-bit, 16x32): lane<16 -> K {0..7,16..23},
// lane>=16 -> K {8..15,24..31}; both as two contiguous 16B chunks at
// row_ptr + kofs and row_ptr + kofs + 16, where kofs = (lane>>4)*8.

// ---- async global->LDS 16B copy (CDNA5 ASYNCcnt path), sync fallback ----
__device__ __forceinline__ void cp16_g2l(const unsigned short* g, unsigned short* l) {
#if __has_builtin(__builtin_amdgcn_global_load_async_to_lds_b128)
  __builtin_amdgcn_global_load_async_to_lds_b128((gv4i*)g, (lv4i*)l, 0, 0);
#else
  *(u32x4*)l = *(const u32x4*)g;          // sync: global_load_b128 + ds_store_b128
#endif
}

__device__ __forceinline__ void wait_async_32() {
#if __has_builtin(__builtin_amdgcn_s_wait_asynccnt)
  __builtin_amdgcn_s_wait_asynccnt(32);
#else
  asm volatile("s_wait_asynccnt 32" ::: "memory");
#endif
  asm volatile("" ::: "memory");
}

__device__ __forceinline__ void wait_async_0() {
#if __has_builtin(__builtin_amdgcn_s_wait_asynccnt)
  __builtin_amdgcn_s_wait_asynccnt(0);
#else
  asm volatile("s_wait_asynccnt 0" ::: "memory");
#endif
  asm volatile("" ::: "memory");
}

// ---------------- fp32 -> bf16 convert ----------------
__global__ void k_f32_to_bf16(const float* __restrict__ in,
                              unsigned short* __restrict__ out, int n) {
  int i = blockIdx.x * 256 + threadIdx.x;
  if (i < n) out[i] = f2bf(in[i]);
}

// ---------------- V -> V^T per head: vt[b][h][d][s] = v[b*S+s][h*HD+d] ----------------
__global__ void k_transpose_v(const unsigned short* __restrict__ vb,
                              unsigned short* __restrict__ vtb) {
  unsigned int i = blockIdx.x * 256 + threadIdx.x;   // over B*NH*HD*SS = 8388608
  int s  = i & (SS - 1);
  int d  = (i >> 11) & (HD - 1);
  int bh = i >> 18;
  int h  = bh & (NH - 1);
  int b  = bh >> 4;
  vtb[i] = vb[((size_t)(b * SS) + s) * DDIM + h * HD + d];
}

// ---------------- GEMM: C[M,N] = A[M,K] @ W[N,K]^T + bias ----------------
// 128 threads (4 waves); block tile 64(M) x 64(N); each wave: 16(M) x 64(N).
// Fragments double-buffered in registers so a full chunk (10 b128 loads) is in
// flight while the previous chunk's 4 WMMAs execute.
__global__ __launch_bounds__(128)
void k_gemm_bf16_wt(const unsigned short* __restrict__ A,
                    const unsigned short* __restrict__ W,
                    const float* __restrict__ bias,
                    unsigned short* __restrict__ outb,
                    float* __restrict__ outf,
                    int M, int N, int K) {
  const int lane = threadIdx.x & 31;
  const int wave = threadIdx.x >> 5;
  const int m0   = blockIdx.y * 64 + wave * 16;
  const int n0   = blockIdx.x * 64;
  const int rl   = lane & 15;
  const int hf   = lane >> 4;
  const int kofs = hf * 8;

  const v8f vzero = {0.f, 0.f, 0.f, 0.f, 0.f, 0.f, 0.f, 0.f};
  v8f acc[4];
#pragma unroll
  for (int j = 0; j < 4; ++j) acc[j] = vzero;

  const unsigned short* arow = A + (size_t)(m0 + rl) * K + kofs;
  const unsigned short* wrow = W + (size_t)(n0 + rl) * K + kofs;

  auto ldA = [&](Frag16& f, int kc) {
    const u32x4* p = (const u32x4*)(arow + kc);
    f.q[0] = p[0]; f.q[1] = p[2];
  };
  auto ldB = [&](Frag16* f, int kc) {
#pragma unroll
    for (int j = 0; j < 4; ++j) {
      const u32x4* p = (const u32x4*)(wrow + (size_t)j * 16 * K + kc);
      f[j].q[0] = p[0]; f[j].q[1] = p[2];
    }
  };

  Frag16 aE, aO, bE[4], bO[4];
  ldA(aE, 0); ldB(bE, 0);
  const int C = K >> 5;                       // #32-wide chunks (even: 16 or 64)
  for (int c = 0; c < C - 2; c += 2) {
    const int kc1 = (c + 1) << 5, kc2 = (c + 2) << 5;
    __builtin_prefetch(arow + kc2 + 64, 0, 1);     // -> global_prefetch_b8
    ldA(aO, kc1); ldB(bO, kc1);
#pragma unroll
    for (int j = 0; j < 4; ++j) acc[j] = wmma_bf16(aE, bE[j], acc[j]);
    ldA(aE, kc2); ldB(bE, kc2);
#pragma unroll
    for (int j = 0; j < 4; ++j) acc[j] = wmma_bf16(aO, bO[j], acc[j]);
  }
  // epilogue: even buffer holds chunk C-2; fetch and finish chunk C-1
  const int kcl = (C - 1) << 5;
  ldA(aO, kcl); ldB(bO, kcl);
#pragma unroll
  for (int j = 0; j < 4; ++j) acc[j] = wmma_bf16(aE, bE[j], acc[j]);
#pragma unroll
  for (int j = 0; j < 4; ++j) acc[j] = wmma_bf16(aO, bO[j], acc[j]);

  // C layout: lane L, vgpr v -> row m = v + 8*(L>>4), col n = L&15
#pragma unroll
  for (int j = 0; j < 4; ++j) {
    const int n = n0 + j * 16 + rl;
    const float bv = bias ? bias[n] : 0.0f;
#pragma unroll
    for (int v = 0; v < 8; ++v) {
      const int row = m0 + v + 8 * hf;
      const float val = acc[j][v] + bv;
      if (outf) outf[(size_t)row * N + n] = val;
      if (outb) outb[(size_t)row * N + n] = f2bf(val);
    }
  }
}

// ---------------- fused causal flash attention ----------------
// One wave per (b, h, 16-query tile). K-tile (32x128) and Vt-tile (128x32) are
// double-buffered in LDS via async global->LDS copies; while chunk c's
// 16 WMMAs + online softmax run, chunk c+1 streams in on the ASYNC engine.
__global__ __launch_bounds__(32)
void k_mla_attn(const unsigned short* __restrict__ qb,
                const unsigned short* __restrict__ kb,
                const unsigned short* __restrict__ vtb,
                unsigned short* __restrict__ aob) {
  const int lane  = threadIdx.x & 31;
  const int blk   = blockIdx.x;
  const int qt    = blk & 127;          // S/16 = 128 query tiles
  const int bh    = blk >> 7;
  const int h     = bh & (NH - 1);
  const int b     = bh >> 4;
  const int qbase = qt * 16;
  const int rl    = lane & 15;
  const int hf    = lane >> 4;
  const int kofs  = hf * 8;

  __shared__ __align__(16) unsigned short klds[2][32 * HD];   // 2 x 8 KB
  __shared__ __align__(16) unsigned short vlds[2][HD * 32];   // 2 x 8 KB
  __shared__ __align__(16) unsigned short pls[16 * 32];       // 1 KB P tile

  const unsigned short* kg = kb  + (size_t)(b * SS) * DDIM + h * HD; // + (key)*D + col
  const unsigned short* vg = vtb + (size_t)bh * HD * SS;             // + d*S + s

  // stage one 32-key chunk (K tile + Vt tile) into LDS buffer `buf`
  auto stage = [&](int kb0, int buf) {
#pragma unroll
    for (int i = 0; i < 16; ++i) {            // K tile: 32 rows x 128 cols
      const int id = i * 32 + lane;
      const int row = id >> 4, cc = id & 15;  // 16 x 16B chunks per row
      cp16_g2l(kg + (size_t)(kb0 + row) * DDIM + cc * 8,
               &klds[buf][row * HD + cc * 8]);
    }
#pragma unroll
    for (int i = 0; i < 16; ++i) {            // Vt tile: 128 rows x 32 cols
      const int id = i * 32 + lane;
      const int row = id >> 2, cc = id & 3;   // 4 x 16B chunks per row
      cp16_g2l(vg + (size_t)row * SS + kb0 + cc * 8,
               &vlds[buf][row * 32 + cc * 8]);
    }
  };

  // resident Q fragments (16 x 128 bf16 = 4 A-fragments), direct from global
  Frag16 fq[4];
  const unsigned short* qrow = qb + ((size_t)(b * SS) + qbase + rl) * DDIM + h * HD + kofs;
#pragma unroll
  for (int kc = 0; kc < 4; ++kc) {
    const u32x4* p = (const u32x4*)(qrow + kc * 32);
    fq[kc].q[0] = p[0];
    fq[kc].q[1] = p[2];
  }

  const v8f vzero = {0.f, 0.f, 0.f, 0.f, 0.f, 0.f, 0.f, 0.f};
  v8f O[8];
#pragma unroll
  for (int t = 0; t < 8; ++t) O[t] = vzero;
  float m[8], l[8];
#pragma unroll
  for (int v = 0; v < 8; ++v) { m[v] = -1e30f; l[v] = 0.0f; }

  const float scale = 0.08838834764831845f;  // 1/sqrt(HD)
  const int nch = (qbase + 16 + 31) >> 5;    // causal key-chunk bound

  stage(0, 0);                               // prologue: chunk 0 -> buf 0

  for (int c = 0; c < nch; ++c) {
    const int kb0 = c * 32;
    const int cur = c & 1;

    if (c + 1 < nch) {                       // overlap: stream chunk c+1
      stage((c + 1) * 32, cur ^ 1);
      wait_async_32();                       // <=32 outstanding => buf `cur` ready
    } else {
      wait_async_0();
    }

    // ---- scores: 16 queries x 32 keys, fragments from LDS ----
    v8f s0 = vzero, s1 = vzero;
#pragma unroll
    for (int kc = 0; kc < 4; ++kc) {
      Frag16 fk0, fk1;
      const u32x4* p0 = (const u32x4*)&klds[cur][rl * HD + kc * 32 + kofs];
      fk0.q[0] = p0[0]; fk0.q[1] = p0[2];
      const u32x4* p1 = (const u32x4*)&klds[cur][(16 + rl) * HD + kc * 32 + kofs];
      fk1.q[0] = p1[0]; fk1.q[1] = p1[2];
      s0 = wmma_bf16(fq[kc], fk0, s0);
      s1 = wmma_bf16(fq[kc], fk1, s1);
    }

    // ---- online softmax update (row r = v + 8*hf; col = lane&15 per half) ----
    float sc[8];
#pragma unroll
    for (int v = 0; v < 8; ++v) {
      const int qrowi = qbase + v + 8 * hf;
      const int k0 = kb0 + rl;
      const int k1 = k0 + 16;
      float a0 = s0[v] * scale; if (k0 > qrowi) a0 = -1e30f;
      float a1 = s1[v] * scale; if (k1 > qrowi) a1 = -1e30f;
      float mx = fmaxf(a0, a1);
      mx = fmaxf(mx, __shfl_xor(mx, 1, 32));
      mx = fmaxf(mx, __shfl_xor(mx, 2, 32));
      mx = fmaxf(mx, __shfl_xor(mx, 4, 32));
      mx = fmaxf(mx, __shfl_xor(mx, 8, 32));
      const float mn = fmaxf(m[v], mx);
      const float p0 = __expf(a0 - mn);
      const float p1 = __expf(a1 - mn);
      float sum = p0 + p1;
      sum += __shfl_xor(sum, 1, 32);
      sum += __shfl_xor(sum, 2, 32);
      sum += __shfl_xor(sum, 4, 32);
      sum += __shfl_xor(sum, 8, 32);
      sc[v] = __expf(m[v] - mn);
      l[v]  = l[v] * sc[v] + sum;
      m[v]  = mn;
      const int lrow = v + 8 * hf;
      pls[lrow * 32 + rl]      = f2bf(p0);
      pls[lrow * 32 + rl + 16] = f2bf(p1);
    }

    // rescale running output accumulator
#pragma unroll
    for (int t = 0; t < 8; ++t)
#pragma unroll
      for (int v = 0; v < 8; ++v) O[t][v] *= sc[v];

    // P tile LDS writes complete before A-fragment readback (wave-internal)
    asm volatile("s_wait_dscnt 0" ::: "memory");

    Frag16 fp;
    const u32x4* pp = (const u32x4*)&pls[rl * 32 + kofs];
    fp.q[0] = pp[0];
    fp.q[1] = pp[2];

    // ---- O += P @ V : Vt fragments from LDS ----
#pragma unroll
    for (int t = 0; t < 8; ++t) {
      Frag16 fv;
      const u32x4* pv = (const u32x4*)&vlds[cur][(t * 16 + rl) * 32 + kofs];
      fv.q[0] = pv[0];
      fv.q[1] = pv[2];
      O[t] = wmma_bf16(fp, fv, O[t]);
    }
  }

  // ---- normalize and store bf16 into head-concat layout ----
#pragma unroll
  for (int t = 0; t < 8; ++t)
#pragma unroll
    for (int v = 0; v < 8; ++v) {
      const int row = qbase + v + 8 * hf;
      const float val = O[t][v] / l[v];
      aob[((size_t)(b * SS) + row) * DDIM + h * HD + t * 16 + rl] = f2bf(val);
    }
}

// ---------------- launcher ----------------
extern "C" void kernel_launch(void* const* d_in, const int* in_sizes, int n_in,
                              void* d_out, int out_size, void* d_ws, size_t ws_size,
                              hipStream_t stream) {
  (void)in_sizes; (void)n_in; (void)out_size; (void)ws_size;
  const float* x    = (const float*)d_in[0];
  const float* Wq_w = (const float*)d_in[1];
  const float* Wq_b = (const float*)d_in[2];
  const float* Wc_w = (const float*)d_in[3];
  const float* Wc_b = (const float*)d_in[4];
  const float* Wk_w = (const float*)d_in[5];
  const float* Wk_b = (const float*)d_in[6];
  const float* Wv_w = (const float*)d_in[7];
  const float* Wv_b = (const float*)d_in[8];
  const float* Wo_w = (const float*)d_in[9];
  const float* Wo_b = (const float*)d_in[10];
  float* out = (float*)d_out;

  char* ws = (char*)d_ws;
  size_t off = 0;
  auto alloc = [&](size_t bytes) -> char* {
    char* p = ws + off;
    off += (bytes + 255) & ~(size_t)255;
    return p;
  };
  const size_t MS = (size_t)BB * SS;                       // 4096 rows
  unsigned short* xb   = (unsigned short*)alloc(MS * DDIM * 2);   // x bf16
  unsigned short* wqb  = (unsigned short*)alloc((size_t)DDIM * DDIM * 2);
  unsigned short* wcb  = (unsigned short*)alloc((size_t)DL * DDIM * 2);
  unsigned short* wkb  = (unsigned short*)alloc((size_t)DDIM * DL * 2);
  unsigned short* wvb  = (unsigned short*)alloc((size_t)DDIM * DL * 2);
  unsigned short* wob  = (unsigned short*)alloc((size_t)DDIM * DDIM * 2);
  unsigned short* qb   = (unsigned short*)alloc(MS * DDIM * 2);
  unsigned short* latb = (unsigned short*)alloc(MS * DL * 2);
  unsigned short* kbuf = (unsigned short*)alloc(MS * DDIM * 2);
  unsigned short* vbuf = (unsigned short*)alloc(MS * DDIM * 2);
  unsigned short* vtb  = (unsigned short*)alloc(MS * DDIM * 2);
  unsigned short* aob  = (unsigned short*)alloc(MS * DDIM * 2);

  // 1) fp32 -> bf16 conversions
  auto conv = [&](const float* src, unsigned short* dst, int n) {
    k_f32_to_bf16<<<(n + 255) / 256, 256, 0, stream>>>(src, dst, n);
  };
  conv(x,    xb,  (int)(MS * DDIM));
  conv(Wq_w, wqb, DDIM * DDIM);
  conv(Wc_w, wcb, DL * DDIM);
  conv(Wk_w, wkb, DDIM * DL);
  conv(Wv_w, wvb, DDIM * DL);
  conv(Wo_w, wob, DDIM * DDIM);

  // 2) projections: C = A @ W^T + b   (grid = (N/64, M/64), 128 threads)
  auto gemm = [&](const unsigned short* A, const unsigned short* W, const float* bias,
                  unsigned short* ob, float* of, int M, int N, int K) {
    dim3 g(N / 64, M / 64);
    k_gemm_bf16_wt<<<g, 128, 0, stream>>>(A, W, bias, ob, of, M, N, K);
  };
  gemm(xb,   wqb, Wq_b, qb,   nullptr, (int)MS, DDIM, DDIM);  // Q
  gemm(xb,   wcb, Wc_b, latb, nullptr, (int)MS, DL,   DDIM);  // kv latent
  gemm(latb, wkb, Wk_b, kbuf, nullptr, (int)MS, DDIM, DL);    // K
  gemm(latb, wvb, Wv_b, vbuf, nullptr, (int)MS, DDIM, DL);    // V

  // 3) V -> V^T per head
  k_transpose_v<<<(int)(MS * DDIM / 256), 256, 0, stream>>>(vbuf, vtb);

  // 4) fused causal attention (one wave per (b,h,16-query tile))
  k_mla_attn<<<BB * NH * (SS / 16), 32, 0, stream>>>(qb, kbuf, vtb, aob);

  // 5) output projection -> fp32 d_out
  gemm(aob, wob, Wo_b, nullptr, out, (int)MS, DDIM, DDIM);
}
